// PointNextEncoder_34737695490784
// MI455X (gfx1250) — matrix operand order repaired and checked by reading, not compiled, not run on hardware
//
#include <hip/hip_runtime.h>
#include <cstdint>

// ---------------------------------------------------------------------------
// PointNeXt encoder for gfx1250 (MI455X).
// Activations + repacked weights stored as bf16; WMMA bf16 (f32 accumulate)
// with 32x32 per-wave output blocking (2 loads per WMMA); fp32 pre-BN
// tensors for exact two-pass batch statistics.
// ---------------------------------------------------------------------------

typedef __attribute__((ext_vector_type(16))) __bf16 v16bf;
typedef __attribute__((ext_vector_type(8)))  float  v8f;
typedef unsigned short bf16_t;

union Frag { v16bf v; uint4 q[2]; bf16_t u[16]; };

#define WMMA_BF16(A, Bf, C) \
    __builtin_amdgcn_wmma_f32_16x16x32_bf16(false, (A), false, (Bf), (short)0, (C), false, false)

__device__ __forceinline__ bf16_t f2bf(float f) {
    union { float f; unsigned u; } x; x.f = f;
    unsigned r = x.u + 0x7FFFu + ((x.u >> 16) & 1u);   // round-to-nearest-even
    return (bf16_t)(r >> 16);
}
__device__ __forceinline__ float bf2f(bf16_t u) {
    union { unsigned u; float f; } x; x.u = ((unsigned)u) << 16; return x.f;
}

// ---------------------------------------------------------------------------
// Head: x[B*N,4] @ W[4,32] + b  -> bf16 activations
// ---------------------------------------------------------------------------
__global__ void head_kernel(const float* __restrict__ x, const float* __restrict__ W,
                            const float* __restrict__ bias, bf16_t* __restrict__ out,
                            int BN, int Cin, int Cout) {
    long t = (long)blockIdx.x * blockDim.x + threadIdx.x;
    if (t >= (long)BN * Cout) return;
    long r = t / Cout; int c = (int)(t % Cout);
    float s = bias[c];
    for (int i = 0; i < Cin; ++i) s += x[r * Cin + i] * W[i * Cout + c];
    out[t] = f2bf(s);
}

// ---------------------------------------------------------------------------
// FPS: one workgroup per batch; per-iteration argmax via wave32 shuffle
// reduction (2 barriers/iter).  Emits [0, a1, ...] like the reference scan.
// ---------------------------------------------------------------------------
#define FPS_T 1024
__device__ __forceinline__ void wave_argmax(float& v, int& i) {
#pragma unroll
    for (int off = 16; off > 0; off >>= 1) {
        float ov = __shfl_xor(v, off, 32);
        int   oi = __shfl_xor(i, off, 32);
        if (ov > v) { v = ov; i = oi; }
    }
}

__global__ void fps_kernel(const float* __restrict__ p, int N, int M, int* __restrict__ sidx) {
    const int b = blockIdx.x, tid = threadIdx.x;
    const int lane = tid & 31, wv = tid >> 5;
    __shared__ float sv[32];
    __shared__ int   si[32];
    __shared__ int   s_last;
    const float* pb = p + (long)b * N * 3;
    float d[8];
#pragma unroll
    for (int j = 0; j < 8; ++j) d[j] = 1e30f;
    int last = 0;
    if (tid == 0) sidx[(long)b * M] = 0;
    for (int i = 1; i < M; ++i) {
        float lx = pb[last * 3 + 0], ly = pb[last * 3 + 1], lz = pb[last * 3 + 2];
        float best = -1.f; int bestn = 0;
#pragma unroll
        for (int j = 0; j < 8; ++j) {
            int n = j * FPS_T + tid;
            if (n < N) {
                float dx = pb[n * 3 + 0] - lx, dy = pb[n * 3 + 1] - ly, dz = pb[n * 3 + 2] - lz;
                float dd = dx * dx + dy * dy + dz * dz;
                d[j] = fminf(d[j], dd);
                if (d[j] > best) { best = d[j]; bestn = n; }
            }
        }
        wave_argmax(best, bestn);
        if (lane == 0) { sv[wv] = best; si[wv] = bestn; }
        __syncthreads();
        if (wv == 0) {
            float v = sv[lane]; int ix = si[lane];
            wave_argmax(v, ix);
            if (lane == 0) { s_last = ix; sidx[(long)b * M + i] = ix; }
        }
        __syncthreads();
        last = s_last;
    }
}

__global__ void gather_p_kernel(const float* __restrict__ p, const int* __restrict__ sidx,
                                float* __restrict__ out, int B, int N, int M) {
    long t = (long)blockIdx.x * blockDim.x + threadIdx.x;
    if (t >= (long)B * M * 3) return;
    int c = (int)(t % 3); long q = t / 3; int b = (int)(q / M); int m = (int)(q % M);
    out[t] = p[((long)b * N + sidx[(long)b * M + m]) * 3 + c];
}

// ---------------------------------------------------------------------------
// Ball query: one wave32 per query; "first K=32 indices within radius"
// (== reference top_k(-key)), padded with the first hit (or 0).
// ---------------------------------------------------------------------------
__global__ void ballq_kernel(const float* __restrict__ pq, const float* __restrict__ ps,
                             int* __restrict__ nidx, int B, int M, int Ns, float r2) {
    int wid = blockIdx.x * (blockDim.x >> 5) + (threadIdx.x >> 5);
    int lane = threadIdx.x & 31;
    if (wid >= B * M) return;
    int b = wid / M;
    const float* psb = ps + (long)b * Ns * 3;
    float qx = pq[(long)wid * 3 + 0], qy = pq[(long)wid * 3 + 1], qz = pq[(long)wid * 3 + 2];
    int* out = nidx + (long)wid * 32;
    int count = 0, firstIdx = 0;
    for (int base = 0; base < Ns && count < 32; base += 32) {
        int n = base + lane;
        bool in = false;
        if (n < Ns) {
            float dx = psb[n * 3 + 0] - qx, dy = psb[n * 3 + 1] - qy, dz = psb[n * 3 + 2] - qz;
            in = (dx * dx + dy * dy + dz * dz) < r2;
        }
        unsigned mask = __builtin_amdgcn_ballot_w32(in);
        if (mask) {
            if (count == 0) firstIdx = base + __builtin_ctz(mask);
            int pos = count + __builtin_popcount(mask & ((1u << lane) - 1u));
            if (in && pos < 32) out[pos] = n;
            count += __builtin_popcount(mask);
            if (count > 32) count = 32;
        }
    }
    if (lane >= count) out[lane] = (count > 0) ? firstIdx : 0;
}

// ---------------------------------------------------------------------------
// Weight repack kernels (fp32 -> transposed bf16), run per layer.
//  LA: W[(3+Cin) x Cout] -> Wt[Cout x (32+Cin)]; padded rows 3..31 are zero,
//      x rows moved to k>=32 so all x fragment loads are contiguous/aligned.
//  PW: W[Cin x Cout] -> Wt[Cout x Cin].
// ---------------------------------------------------------------------------
__global__ void repack_law_kernel(const float* __restrict__ W, bf16_t* __restrict__ Wt,
                                  int Cin, int Cout) {
    const int Kpad = Cin + 32;
    long t = (long)blockIdx.x * blockDim.x + threadIdx.x;
    if (t >= (long)Cout * Kpad) return;
    int k = (int)(t % Kpad); int c = (int)(t / Kpad);
    float v = 0.f;
    if (k < 3)        v = W[(long)k * Cout + c];
    else if (k >= 32) v = W[(long)(k - 29) * Cout + c];
    Wt[t] = f2bf(v);
}

__global__ void repack_w_kernel(const float* __restrict__ W, bf16_t* __restrict__ Wt,
                                int Cin, int Cout) {
    long t = (long)blockIdx.x * blockDim.x + threadIdx.x;
    if (t >= (long)Cout * Cin) return;
    int k = (int)(t % Cin); int c = (int)(t / Cin);
    Wt[t] = f2bf(W[(long)k * Cout + c]);
}

// ---------------------------------------------------------------------------
// Grouped local-aggregation GEMM (WMMA bf16):
//   Y[b,m,k,:] = concat(dp, x[idx]) @ W
// One wave computes a 32(neighbors) x 32(channels) output block:
// per k-tile 4 A-loads + 4 B-loads (b128) -> 4 WMMAs.
// ---------------------------------------------------------------------------
__global__ void la_wmma_kernel(const float* __restrict__ pq, const float* __restrict__ ps,
                               const bf16_t* __restrict__ xs, const int* __restrict__ nidx,
                               const bf16_t* __restrict__ Wt, float* __restrict__ Y,
                               int B, int M, int Ns, int Cin, int Cout) {
    const int lane = threadIdx.x & 31;
    const int ngroups = Cout >> 5;                    // 32-column groups
    int wid = blockIdx.x * (blockDim.x >> 5) + (threadIdx.x >> 5);
    if (wid >= B * M * ngroups) return;
    const int ng = wid % ngroups;
    const long q = wid / ngroups;
    const int b = (int)(q / M);
    const int Kpad = Cin + 32;
    const int ktiles = Kpad >> 5;
    const int lhi = lane >> 4, llo = lane & 15;
    const long bNs = (long)b * Ns;

    const int* qn = nidx + q * 32;
    const int n0 = qn[llo];                           // A rows 0..15 (lane row = llo)
    const int n1 = qn[16 + llo];                      // A rows 16..31
    const int c0 = ng * 32 + llo;
    const bf16_t* w0 = Wt + (long)c0 * Kpad + lhi * 8;
    const bf16_t* w1 = w0 + (long)16 * Kpad;
    const bf16_t* x0 = xs + (bNs + n0) * Cin + lhi * 8;
    const bf16_t* x1 = xs + (bNs + n1) * Cin + lhi * 8;

    Frag a0, a1, b0, b1;
    // k-tile 0: [dp0 dp1 dp2 0 ... 0]; only lane-half 0 carries nonzeros.
    a0.q[0] = make_uint4(0, 0, 0, 0); a0.q[1] = make_uint4(0, 0, 0, 0);
    a1.q[0] = make_uint4(0, 0, 0, 0); a1.q[1] = make_uint4(0, 0, 0, 0);
    if (lhi == 0) {
        float qx = pq[q * 3 + 0], qy = pq[q * 3 + 1], qz = pq[q * 3 + 2];
        a0.u[0] = f2bf(ps[(bNs + n0) * 3 + 0] - qx);
        a0.u[1] = f2bf(ps[(bNs + n0) * 3 + 1] - qy);
        a0.u[2] = f2bf(ps[(bNs + n0) * 3 + 2] - qz);
        a1.u[0] = f2bf(ps[(bNs + n1) * 3 + 0] - qx);
        a1.u[1] = f2bf(ps[(bNs + n1) * 3 + 1] - qy);
        a1.u[2] = f2bf(ps[(bNs + n1) * 3 + 2] - qz);
    }
    b0.q[0] = *(const uint4*)(w0); b0.q[1] = *(const uint4*)(w0 + 16);
    b1.q[0] = *(const uint4*)(w1); b1.q[1] = *(const uint4*)(w1 + 16);
    w0 += 32; w1 += 32;

    v8f acc00 = {}, acc01 = {}, acc10 = {}, acc11 = {};
    acc00 = WMMA_BF16(a0.v, b0.v, acc00);
    acc10 = WMMA_BF16(a1.v, b0.v, acc10);
    acc01 = WMMA_BF16(a0.v, b1.v, acc01);
    acc11 = WMMA_BF16(a1.v, b1.v, acc11);

    for (int kt = 1; kt < ktiles; ++kt) {
        a0.q[0] = *(const uint4*)(x0); a0.q[1] = *(const uint4*)(x0 + 16);
        a1.q[0] = *(const uint4*)(x1); a1.q[1] = *(const uint4*)(x1 + 16);
        b0.q[0] = *(const uint4*)(w0); b0.q[1] = *(const uint4*)(w0 + 16);
        b1.q[0] = *(const uint4*)(w1); b1.q[1] = *(const uint4*)(w1 + 16);
        x0 += 32; x1 += 32; w0 += 32; w1 += 32;
        acc00 = WMMA_BF16(a0.v, b0.v, acc00);
        acc10 = WMMA_BF16(a1.v, b0.v, acc10);
        acc01 = WMMA_BF16(a0.v, b1.v, acc01);
        acc11 = WMMA_BF16(a1.v, b1.v, acc11);
    }

    float* yq = Y + q * 32 * Cout;
#pragma unroll
    for (int r = 0; r < 8; ++r) {                     // D: VGPR r -> row r + 8*lhi
        int row0 = r + 8 * lhi, row1 = 16 + row0;
        yq[(long)row0 * Cout + c0]      = acc00[r];
        yq[(long)row0 * Cout + c0 + 16] = acc01[r];
        yq[(long)row1 * Cout + c0]      = acc10[r];
        yq[(long)row1 * Cout + c0 + 16] = acc11[r];
    }
}

// ---------------------------------------------------------------------------
// Dense GEMM (WMMA bf16): Y[R,Cout](fp32) = X[R,Cin](bf16) @ Wt^T
// One wave computes a 32x32 output block: 4+4 b128 loads -> 4 WMMAs per tile.
// ---------------------------------------------------------------------------
__global__ void gemm_wmma_kernel(const bf16_t* __restrict__ X, const bf16_t* __restrict__ Wt,
                                 float* __restrict__ Y, int R, int Cin, int Cout) {
    const int lane = threadIdx.x & 31;
    const int ngroups = Cout >> 5;
    const int rgroups = R >> 5;
    int wid = blockIdx.x * (blockDim.x >> 5) + (threadIdx.x >> 5);
    if (wid >= rgroups * ngroups) return;
    const int ng = wid % ngroups;
    const int rg = wid / ngroups;
    const int lhi = lane >> 4, llo = lane & 15;
    const int c0 = ng * 32 + llo;
    const long row0 = (long)rg * 32 + llo;
    const bf16_t* x0 = X + row0 * Cin + lhi * 8;
    const bf16_t* x1 = X + (row0 + 16) * Cin + lhi * 8;
    const bf16_t* w0 = Wt + (long)c0 * Cin + lhi * 8;
    const bf16_t* w1 = w0 + (long)16 * Cin;
    const int ktiles = Cin >> 5;

    Frag a0, a1, b0, b1;
    v8f acc00 = {}, acc01 = {}, acc10 = {}, acc11 = {};
    for (int kt = 0; kt < ktiles; ++kt) {
        a0.q[0] = *(const uint4*)(x0); a0.q[1] = *(const uint4*)(x0 + 16);
        a1.q[0] = *(const uint4*)(x1); a1.q[1] = *(const uint4*)(x1 + 16);
        b0.q[0] = *(const uint4*)(w0); b0.q[1] = *(const uint4*)(w0 + 16);
        b1.q[0] = *(const uint4*)(w1); b1.q[1] = *(const uint4*)(w1 + 16);
        x0 += 32; x1 += 32; w0 += 32; w1 += 32;
        acc00 = WMMA_BF16(a0.v, b0.v, acc00);
        acc10 = WMMA_BF16(a1.v, b0.v, acc10);
        acc01 = WMMA_BF16(a0.v, b1.v, acc01);
        acc11 = WMMA_BF16(a1.v, b1.v, acc11);
    }
    float* yr = Y + (long)rg * 32 * Cout;
#pragma unroll
    for (int r = 0; r < 8; ++r) {
        int rr0 = r + 8 * lhi, rr1 = 16 + rr0;
        yr[(long)rr0 * Cout + c0]      = acc00[r];
        yr[(long)rr0 * Cout + c0 + 16] = acc01[r];
        yr[(long)rr1 * Cout + c0]      = acc10[r];
        yr[(long)rr1 * Cout + c0 + 16] = acc11[r];
    }
}

// ---------------------------------------------------------------------------
// Per-channel batch stats: one block per channel, no atomics.
// ---------------------------------------------------------------------------
__global__ void stats_kernel(const float* __restrict__ Y, long rows, int C,
                             float* __restrict__ mean, float* __restrict__ rstd) {
    const int c = blockIdx.x, tid = threadIdx.x;
    __shared__ float s1[256], s2[256];
    float a = 0.f, qq = 0.f;
    for (long r = tid; r < rows; r += 256) {
        float y = Y[r * C + c];
        a += y; qq += y * y;
    }
    s1[tid] = a; s2[tid] = qq;
    __syncthreads();
    for (int s = 128; s > 0; s >>= 1) {
        if (tid < s) { s1[tid] += s1[tid + s]; s2[tid] += s2[tid + s]; }
        __syncthreads();
    }
    if (tid == 0) {
        float m = s1[0] / (float)rows;
        float v = s2[0] / (float)rows - m * m;
        mean[c] = m; rstd[c] = rsqrtf(v + 1e-5f);
    }
}

// BN + ReLU + max over K=32 neighbor slots -> bf16
__global__ void bn_relu_max_kernel(const float* __restrict__ Y, const float* __restrict__ mean,
                                   const float* __restrict__ rstd, const float* __restrict__ g,
                                   const float* __restrict__ bb, bf16_t* __restrict__ out,
                                   long Q, int C) {
    long t = (long)blockIdx.x * blockDim.x + threadIdx.x;
    if (t >= Q * C) return;
    long q = t / C; int c = (int)(t % C);
    float gm = g[c] * rstd[c];
    float bm = bb[c] - gm * mean[c];
    const float* yq = Y + q * 32 * C;
    float mx = 0.f;                        // relu >= 0, so 0 is the max identity
#pragma unroll 4
    for (int k = 0; k < 32; ++k) {
        float v = gm * yq[(long)k * C + c] + bm;
        mx = fmaxf(mx, fmaxf(v, 0.f));
    }
    out[t] = f2bf(mx);
}

// BN + ReLU -> bf16
__global__ void bn_relu_kernel(const float* __restrict__ Y, const float* __restrict__ mean,
                               const float* __restrict__ rstd, const float* __restrict__ g,
                               const float* __restrict__ bb, bf16_t* __restrict__ out,
                               long R, int C) {
    long t = (long)blockIdx.x * blockDim.x + threadIdx.x;
    if (t >= R * C) return;
    int c = (int)(t % C);
    float v = g[c] * (Y[t] - mean[c]) * rstd[c] + bb[c];
    out[t] = f2bf(fmaxf(v, 0.f));
}

// BN (no act) + bf16 residual + ReLU -> bf16
__global__ void bn_add_relu_kernel(const float* __restrict__ Y, const float* __restrict__ mean,
                                   const float* __restrict__ rstd, const float* __restrict__ g,
                                   const float* __restrict__ bb, const bf16_t* __restrict__ xres,
                                   bf16_t* __restrict__ out, long R, int C) {
    long t = (long)blockIdx.x * blockDim.x + threadIdx.x;
    if (t >= R * C) return;
    int c = (int)(t % C);
    float v = g[c] * (Y[t] - mean[c]) * rstd[c] + bb[c] + bf2f(xres[t]);
    out[t] = f2bf(fmaxf(v, 0.f));
}

__global__ void pack_out_kernel(const float* __restrict__ p, const bf16_t* __restrict__ x,
                                float* __restrict__ out, int np, int nx) {
    int t = blockIdx.x * blockDim.x + threadIdx.x;
    if (t < np) out[t] = p[t];
    else if (t < np + nx) out[t] = bf2f(x[t - np]);
}

// ---------------------------------------------------------------------------
// Orchestration
// ---------------------------------------------------------------------------
extern "C" void kernel_launch(void* const* d_in, const int* in_sizes, int n_in,
                              void* d_out, int out_size, void* d_ws, size_t ws_size,
                              hipStream_t stream) {
    (void)in_sizes; (void)n_in; (void)out_size; (void)ws_size;
    const int B = 4, N0 = 8192, CIN = 4, K = 32;
    const float SA_R[4]  = {0.1f, 0.2f, 0.4f, 0.8f};
    const float BLK_R[4] = {0.2f, 0.4f, 0.8f, 1.6f};
    const int CH[5] = {32, 64, 128, 256, 512};
    const int NB[4] = {3, 6, 3, 3};

    const float* d_p = (const float*)d_in[0];
    const float* d_x = (const float*)d_in[1];

    // Param decode (jax sorted-key pytree flatten):
    //   head{W,b}; per stage: blocks[{la_W,la_b,la_g,pw1_W,pw1_b,pw1_g,pw2_W,pw2_b,pw2_g}], sa{W,b,g}
    int pi = 2;
    const float* headW = (const float*)d_in[pi++];
    const float* headB = (const float*)d_in[pi++];
    const float* blkP[4][6][9];
    const float* saP[4][3];
    for (int si = 0; si < 4; ++si) {
        for (int bk = 0; bk < NB[si]; ++bk)
            for (int f = 0; f < 9; ++f) blkP[si][bk][f] = (const float*)d_in[pi++];
        for (int f = 0; f < 3; ++f) saP[si][f] = (const float*)d_in[pi++];
    }

    // Workspace carve-out
    char* w = (char*)d_ws;
    auto alloc = [&](size_t nbytes) -> void* {
        void* r = (void*)w;
        w += (nbytes + 255) & ~(size_t)255;
        return r;
    };
    float*  pbuf[2] = { (float*)alloc((size_t)B * N0 * 3 * 4), (float*)alloc((size_t)B * N0 * 3 * 4) };
    const size_t XN = (size_t)B * N0 * 32;                 // max activation element count
    bf16_t* xb[2] = { (bf16_t*)alloc(XN * 2), (bf16_t*)alloc(XN * 2) };
    bf16_t* bufA  = (bf16_t*)alloc(XN * 2);                // LA max-pool output (GEMM input)
    float*  bufB  = (float*)alloc(XN * 4);                 // pw2 pre-BN output
    float*  Ybuf  = (float*)alloc((size_t)B * 2048 * 32 * 64 * 4);   // max pre-BN grouped tensor
    float*  Y2buf = (float*)alloc((size_t)8192 * 256 * 4);           // pw1 pre-BN output
    bf16_t* Y2b   = (bf16_t*)alloc((size_t)8192 * 256 * 2);          // pw1 post-BN (pw2 input)
    bf16_t* Wt    = (bf16_t*)alloc((size_t)2048 * 1024 * 2);         // repacked weights
    int*    sidx  = (int*)alloc((size_t)B * 2048 * 4);
    int*    nidx  = (int*)alloc((size_t)B * 2048 * 32 * 4);
    float*  meanb = (float*)alloc(2048 * 4);
    float*  rstdb = (float*)alloc(2048 * 4);

    // Head
    {
        long tot = (long)B * N0 * CH[0];
        head_kernel<<<(int)((tot + 255) / 256), 256, 0, stream>>>(d_x, headW, headB, xb[0], B * N0, CIN, CH[0]);
    }

    const float* p_src = d_p;
    int Ns = N0, pcur = 0;
    bf16_t* x_cur = xb[0];
    bf16_t* x_nxt = xb[1];

    for (int si = 0; si < 4; ++si) {
        const int M = Ns / 4, Cin = CH[si], Cout = CH[si + 1];
        const long R = (long)B * M;

        // --- SA transition ---
        float* p_new = pbuf[pcur]; pcur ^= 1;
        fps_kernel<<<B, FPS_T, 0, stream>>>(p_src, Ns, M, sidx);
        gather_p_kernel<<<(int)((R * 3 + 255) / 256), 256, 0, stream>>>(p_src, sidx, p_new, B, Ns, M);
        float r = SA_R[si];
        ballq_kernel<<<(B * M + 3) / 4, 128, 0, stream>>>(p_new, p_src, nidx, B, M, Ns, r * r);
        {
            long wn = (long)Cout * (Cin + 32);
            repack_law_kernel<<<(int)((wn + 255) / 256), 256, 0, stream>>>(saP[si][0], Wt, Cin, Cout);
        }
        int waves = B * M * (Cout / 32);
        la_wmma_kernel<<<(waves + 3) / 4, 128, 0, stream>>>(p_new, p_src, x_cur, nidx, Wt,
                                                            Ybuf, B, M, Ns, Cin, Cout);
        stats_kernel<<<Cout, 256, 0, stream>>>(Ybuf, R * K, Cout, meanb, rstdb);
        bn_relu_max_kernel<<<(int)((R * Cout + 255) / 256), 256, 0, stream>>>(
            Ybuf, meanb, rstdb, saP[si][2], saP[si][1], x_nxt, R, Cout);
        { bf16_t* t = x_cur; x_cur = x_nxt; x_nxt = t; }
        p_src = p_new;
        Ns = M;

        // --- Inverted-residual blocks ---
        for (int bk = 0; bk < NB[si]; ++bk) {
            const float* const* P = blkP[si][bk];
            float rb = BLK_R[si];
            ballq_kernel<<<(B * M + 3) / 4, 128, 0, stream>>>(p_src, p_src, nidx, B, M, M, rb * rb);
            {
                long wn = (long)Cout * (Cout + 32);
                repack_law_kernel<<<(int)((wn + 255) / 256), 256, 0, stream>>>(P[0], Wt, Cout, Cout);
            }
            la_wmma_kernel<<<(B * M * (Cout / 32) + 3) / 4, 128, 0, stream>>>(
                p_src, p_src, x_cur, nidx, Wt, Ybuf, B, M, M, Cout, Cout);
            stats_kernel<<<Cout, 256, 0, stream>>>(Ybuf, R * K, Cout, meanb, rstdb);
            bn_relu_max_kernel<<<(int)((R * Cout + 255) / 256), 256, 0, stream>>>(
                Ybuf, meanb, rstdb, P[2], P[1], bufA, R, Cout);

            const int C1 = 4 * Cout;
            {
                long wn = (long)C1 * Cout;
                repack_w_kernel<<<(int)((wn + 255) / 256), 256, 0, stream>>>(P[3], Wt, Cout, C1);
            }
            gemm_wmma_kernel<<<(int)(((R / 32) * (C1 / 32) + 3) / 4), 128, 0, stream>>>(
                bufA, Wt, Y2buf, (int)R, Cout, C1);
            stats_kernel<<<C1, 256, 0, stream>>>(Y2buf, R, C1, meanb, rstdb);
            bn_relu_kernel<<<(int)((R * C1 + 255) / 256), 256, 0, stream>>>(
                Y2buf, meanb, rstdb, P[5], P[4], Y2b, R, C1);

            {
                long wn = (long)Cout * C1;
                repack_w_kernel<<<(int)((wn + 255) / 256), 256, 0, stream>>>(P[6], Wt, C1, Cout);
            }
            gemm_wmma_kernel<<<(int)(((R / 32) * (Cout / 32) + 3) / 4), 128, 0, stream>>>(
                Y2b, Wt, bufB, (int)R, C1, Cout);
            stats_kernel<<<Cout, 256, 0, stream>>>(bufB, R, Cout, meanb, rstdb);
            bn_add_relu_kernel<<<(int)((R * Cout + 255) / 256), 256, 0, stream>>>(
                bufB, meanb, rstdb, P[8], P[7], x_cur, x_nxt, R, Cout);
            { bf16_t* t = x_cur; x_cur = x_nxt; x_nxt = t; }
        }
    }

    // Final (p, x) packed flat: p [4,32,3] then x [4,32,512]
    const int np = B * 32 * 3, nx = B * 32 * CH[4];
    pack_out_kernel<<<(np + nx + 255) / 256, 256, 0, stream>>>(p_src, x_cur, (float*)d_out, np, nx);
}
